// CrossAttention_5403068858931
// MI455X (gfx1250) — compile-verified, matrix operand order checked by reading, not run on hardware
//
#include <hip/hip_runtime.h>
#include <hip/hip_bf16.h>

#define DIMX 512
#define HEADSX 8
#define BX 32
#define SX 4096

typedef __attribute__((ext_vector_type(16))) __bf16 v16bf;
typedef __attribute__((ext_vector_type(8)))  __bf16 v8bf;
typedef __attribute__((ext_vector_type(8)))  float  v8f;

// ---------------------------------------------------------------------------
// K0: Qfull = q @ wq^T + bq ; G[b][h][c] = sum_d wk[h*64+d][c] * Qfull[h*64+d]
// ---------------------------------------------------------------------------
__global__ void __launch_bounds__(256)
ca_qg_kernel(const float* __restrict__ q, const float* __restrict__ w_in,
             const float* __restrict__ b_in, float* __restrict__ G) {
  const int b = blockIdx.x;
  const int t = threadIdx.x;
  __shared__ float qs[DIMX];
  __shared__ float Qf[DIMX];
  qs[t]       = q[(size_t)b * DIMX + t];
  qs[t + 256] = q[(size_t)b * DIMX + t + 256];
  __syncthreads();
  #pragma unroll
  for (int rep = 0; rep < 2; ++rep) {
    const int j = t + rep * 256;
    const float* wr = w_in + (size_t)j * DIMX;   // wq rows 0..511
    float acc = b_in[j];
    for (int c = 0; c < DIMX; ++c) acc += qs[c] * wr[c];
    Qf[j] = acc;
  }
  __syncthreads();
  // G: per head, coalesced stream of wk rows (rows 512..1023 of in_proj_w)
  for (int h = 0; h < HEADSX; ++h) {
    float a0 = 0.f, a1 = 0.f;
    for (int d = 0; d < 64; ++d) {
      const int j = h * 64 + d;
      const float qv = Qf[j];
      const float* wr = w_in + (size_t)(DIMX + j) * DIMX;
      a0 += wr[t] * qv;
      a1 += wr[t + 256] * qv;
    }
    G[((size_t)b * HEADSX + h) * DIMX + t]       = a0;
    G[((size_t)b * HEADSX + h) * DIMX + t + 256] = a1;
  }
}

// ---------------------------------------------------------------------------
// K1: scores[b][s][h] = k[b][s][:] . G[b][h][:]   (WMMA bf16, heads in N dim)
// grid = 32 b * 16 s-chunks ; 256 thr = 8 waves ; wave handles 2x16 s rows
// ---------------------------------------------------------------------------
__global__ void __launch_bounds__(256)
ca_scores_kernel(const float* __restrict__ k, const float* __restrict__ G,
                 float* __restrict__ SC) {
  const int b     = blockIdx.x >> 4;
  const int sbase = (blockIdx.x & 15) * 256;
  const int t     = threadIdx.x;
  const int wave  = t >> 5;
  const int lane  = t & 31;
  const int half  = lane >> 4;
  const int l15   = lane & 15;
  const int GP    = 520;  // bf16 pitch: 1040B rows -> N-lanes land on distinct banks

  __shared__ __align__(16) __bf16 g_lds[16 * 520];
  for (int e = t; e < 16 * GP; e += 256) {
    const int h = e / GP, c = e % GP;
    const float val = (h < HEADSX && c < DIMX)
                        ? G[((size_t)b * HEADSX + h) * DIMX + c] : 0.f;
    g_lds[e] = (__bf16)val;
  }
  __syncthreads();

  v8f acc0 = {}, acc1 = {};
  const int s0 = sbase + wave * 32;
  const float* k0r = k + ((size_t)b * SX + s0 + l15) * DIMX;        // tile 0 row
  const float* k1r = k0r + (size_t)16 * DIMX;                        // tile 1 row
  const __bf16* grow = g_lds + l15 * GP;                             // B: N = l15

  #pragma unroll 4
  for (int ks = 0; ks < 16; ++ks) {
    const int r1 = ks * 32 + half * 8;   // K runs {r1..r1+7, r1+16..r1+23}
    // B fragment (G in LDS, bf16, 2x ds_load_b128)
    v8bf bg1 = *(const v8bf*)(grow + r1);
    v8bf bg2 = *(const v8bf*)(grow + r1 + 16);
    v16bf bfrag;
    #pragma unroll
    for (int e = 0; e < 8; ++e) { bfrag[e] = bg1[e]; bfrag[e + 8] = bg2[e]; }
    // A fragments (k rows, f32 -> bf16)
    const float4* p0 = (const float4*)(k0r + r1);
    const float4* p1 = (const float4*)(k1r + r1);
    float4 x0 = p0[0], x1 = p0[1], x2 = p0[4], x3 = p0[5];
    float4 y0 = p1[0], y1 = p1[1], y2 = p1[4], y3 = p1[5];
    float ta[16] = {x0.x,x0.y,x0.z,x0.w, x1.x,x1.y,x1.z,x1.w,
                    x2.x,x2.y,x2.z,x2.w, x3.x,x3.y,x3.z,x3.w};
    float tb[16] = {y0.x,y0.y,y0.z,y0.w, y1.x,y1.y,y1.z,y1.w,
                    y2.x,y2.y,y2.z,y2.w, y3.x,y3.y,y3.z,y3.w};
    v16bf a0, a1;
    #pragma unroll
    for (int e = 0; e < 16; ++e) { a0[e] = (__bf16)ta[e]; a1[e] = (__bf16)tb[e]; }
    acc0 = __builtin_amdgcn_wmma_f32_16x16x32_bf16(false, a0, false, bfrag,
                                                   (short)0, acc0, false, false);
    acc1 = __builtin_amdgcn_wmma_f32_16x16x32_bf16(false, a1, false, bfrag,
                                                   (short)0, acc1, false, false);
  }
  // C layout: VGPR i -> M = i + 8*half ; N = l15 (head, valid < 8)
  if (l15 < HEADSX) {
    const int h = l15;
    #pragma unroll
    for (int i = 0; i < 8; ++i) {
      const int sa = s0 + half * 8 + i;
      SC[((size_t)b * SX + sa) * HEADSX + h]      = acc0[i];
      SC[((size_t)b * SX + sa + 16) * HEADSX + h] = acc1[i];
    }
  }
}

// ---------------------------------------------------------------------------
// K2: softmax over s per (b,h); scale 1/sqrt(64)=0.125 folded in here
// ---------------------------------------------------------------------------
__global__ void __launch_bounds__(256)
ca_softmax_kernel(const float* __restrict__ SC, float* __restrict__ AT) {
  const int b = blockIdx.x >> 3, h = blockIdx.x & 7;
  const int t = threadIdx.x;
  __shared__ float red[256];
  float x[16];
  float m = -1e30f;
  #pragma unroll
  for (int i = 0; i < 16; ++i) {
    const int s = i * 256 + t;
    x[i] = SC[((size_t)b * SX + s) * HEADSX + h] * 0.125f;
    m = fmaxf(m, x[i]);
  }
  red[t] = m; __syncthreads();
  for (int o = 128; o > 0; o >>= 1) {
    if (t < o) red[t] = fmaxf(red[t], red[t + o]);
    __syncthreads();
  }
  m = red[0]; __syncthreads();
  float sum = 0.f;
  #pragma unroll
  for (int i = 0; i < 16; ++i) { x[i] = __expf(x[i] - m); sum += x[i]; }
  red[t] = sum; __syncthreads();
  for (int o = 128; o > 0; o >>= 1) {
    if (t < o) red[t] += red[t + o];
    __syncthreads();
  }
  const float inv = 1.f / red[0];
  #pragma unroll
  for (int i = 0; i < 16; ++i)
    AT[((size_t)b * HEADSX + h) * SX + i * 256 + t] = x[i] * inv;
}

// ---------------------------------------------------------------------------
// K3: w[b][s] = mean_h attn[b][h][s]  (second output region)
// ---------------------------------------------------------------------------
__global__ void __launch_bounds__(256)
ca_wmean_kernel(const float* __restrict__ AT, float* __restrict__ wout) {
  const int b = blockIdx.x >> 4;
  const int s = (blockIdx.x & 15) * 256 + threadIdx.x;
  float acc = 0.f;
  #pragma unroll
  for (int h = 0; h < HEADSX; ++h) acc += AT[((size_t)b * HEADSX + h) * SX + s];
  wout[(size_t)b * SX + s] = acc * (1.f / HEADSX);
}

// ---------------------------------------------------------------------------
// K4: U[b][h][c] = sum_s attn[b][h][s] * v[b][s][c]   (WMMA bf16)
// grid = 32 b * 8 col-chunks(64) ; 128 thr = 4 waves, wave owns 16 cols.
// v is staged 64(rows) x 64(cols) per step into a TRANSPOSED bf16 LDS tile:
//   global: column-per-thread dword loads (lane index = column -> coalesced)
//   LDS   : contiguous ds_store_b128 runs; B fragments = 2x ds_load_b128.
// Double-buffered so the next tile's loads overlap this tile's WMMAs.
// ---------------------------------------------------------------------------
#define CTP 72   // transposed-tile pitch in bf16 elements (144B rows)

__device__ __forceinline__ void ca_stage_tile(const float* __restrict__ src,
                                              __bf16* __restrict__ dst) {
  // src: &v[row0][col_s] (stride DIMX) ; dst: &vT[col_s*CTP + row0_local]
  float cv[32];
  #pragma unroll
  for (int r = 0; r < 32; ++r) cv[r] = src[(size_t)r * DIMX];
  #pragma unroll
  for (int rr = 0; rr < 4; ++rr) {
    v8bf pk;
    #pragma unroll
    for (int e = 0; e < 8; ++e) pk[e] = (__bf16)cv[rr * 8 + e];
    *(v8bf*)(dst + rr * 8) = pk;
  }
}

__global__ void __launch_bounds__(128)
ca_ctx_kernel(const float* __restrict__ v, const float* __restrict__ AT,
              float* __restrict__ U) {
  const int b     = blockIdx.x >> 3;
  const int cbase = (blockIdx.x & 7) * 64;
  const int t     = threadIdx.x;
  const int wave  = t >> 5;
  const int lane  = t & 31;
  const int half  = lane >> 4;
  const int l15   = lane & 15;
  const int col   = wave * 16 + l15;      // this lane's column within the chunk

  // staging role: column col_s, row group g (rows g*32 .. g*32+31 of each tile)
  const int col_s = t & 63;
  const int g     = t >> 6;

  __shared__ __align__(16) __bf16 vT[2][64 * CTP];

  v8f acc = {};
  const float* arow = AT + ((size_t)b * HEADSX + l15) * SX;  // A: M row = head
  const bool hv = (l15 < HEADSX);
  const float* vstage = v + (size_t)b * SX * DIMX + cbase + col_s
                          + (size_t)(g * 32) * DIMX;
  __bf16* const dst0 = &vT[0][col_s * CTP + g * 32];
  __bf16* const dst1 = &vT[1][col_s * CTP + g * 32];

  const int NT = SX / 64;                  // 64 tiles of 64 k-rows
  ca_stage_tile(vstage, dst0);
  for (int it = 0; it < NT; ++it) {
    __syncthreads();                       // staged tile visible / WAR safe
    if (it + 1 < NT)
      ca_stage_tile(vstage + (size_t)(it + 1) * 64 * DIMX,
                    ((it + 1) & 1) ? dst1 : dst0);
    if (it + 2 < NT)                       // L2 warm hint, two tiles ahead
      __builtin_prefetch(vstage + (size_t)(it + 2) * 64 * DIMX, 0, 1);

    const __bf16* buf = vT[it & 1];
    const int k0 = it * 64;
    #pragma unroll
    for (int kk = 0; kk < 64; kk += 32) {
      // A fragment: attn rows (contiguous f32 -> bf16)
      v16bf afrag;
      if (hv) {
        const float4* ap = (const float4*)(arow + k0 + kk + half * 8);
        float4 y0 = ap[0], y1 = ap[1], y2 = ap[4], y3 = ap[5];
        float ta[16] = {y0.x,y0.y,y0.z,y0.w, y1.x,y1.y,y1.z,y1.w,
                        y2.x,y2.y,y2.z,y2.w, y3.x,y3.y,y3.z,y3.w};
        #pragma unroll
        for (int e = 0; e < 16; ++e) afrag[e] = (__bf16)ta[e];
      } else {
        #pragma unroll
        for (int e = 0; e < 16; ++e) afrag[e] = (__bf16)0.f;
      }
      // B fragment: transposed LDS tile, two contiguous 16B runs
      const __bf16* bp = buf + col * CTP + kk + half * 8;
      v8bf b1 = *(const v8bf*)bp;
      v8bf b2 = *(const v8bf*)(bp + 16);
      v16bf bfrag;
      #pragma unroll
      for (int e = 0; e < 8; ++e) { bfrag[e] = b1[e]; bfrag[e + 8] = b2[e]; }
      acc = __builtin_amdgcn_wmma_f32_16x16x32_bf16(false, afrag, false, bfrag,
                                                    (short)0, acc, false, false);
    }
  }
  // C: lanes 0-15 hold M = i (= head i), N = col
  if (half == 0) {
    #pragma unroll
    for (int i = 0; i < 8; ++i)
      U[((size_t)b * HEADSX + i) * DIMX + cbase + col] = acc[i];
  }
}

// ---------------------------------------------------------------------------
// K5: ctx = U @ wv^T + bv ; z = ctx @ out_w^T + out_b  (first output region)
// ---------------------------------------------------------------------------
__global__ void __launch_bounds__(256)
ca_outproj_kernel(const float* __restrict__ U, const float* __restrict__ w_in,
                  const float* __restrict__ b_in, const float* __restrict__ ow,
                  const float* __restrict__ ob, float* __restrict__ zout) {
  const int b = blockIdx.x, t = threadIdx.x;
  __shared__ float ctx[DIMX];
  #pragma unroll
  for (int rep = 0; rep < 2; ++rep) {
    const int j = t + rep * 256;
    const int h = j >> 6;
    const float* ur = U + ((size_t)b * HEADSX + h) * DIMX;
    const float* wr = w_in + (size_t)(2 * DIMX + j) * DIMX;   // wv rows
    float a = b_in[2 * DIMX + j];
    for (int c = 0; c < DIMX; ++c) a += ur[c] * wr[c];
    ctx[j] = a;
  }
  __syncthreads();
  #pragma unroll
  for (int rep = 0; rep < 2; ++rep) {
    const int j = t + rep * 256;
    const float* wr = ow + (size_t)j * DIMX;
    float a = ob[j];
    for (int c = 0; c < DIMX; ++c) a += ctx[c] * wr[c];
    zout[(size_t)b * DIMX + j] = a;
  }
}

// ---------------------------------------------------------------------------
extern "C" void kernel_launch(void* const* d_in, const int* in_sizes, int n_in,
                              void* d_out, int out_size, void* d_ws, size_t ws_size,
                              hipStream_t stream) {
  (void)in_sizes; (void)n_in; (void)out_size; (void)ws_size;
  const float* q    = (const float*)d_in[0];
  const float* k    = (const float*)d_in[1];
  const float* v    = (const float*)d_in[2];
  const float* w_in = (const float*)d_in[3];
  const float* b_in = (const float*)d_in[4];
  const float* ow   = (const float*)d_in[5];
  const float* ob   = (const float*)d_in[6];
  float* out = (float*)d_out;            // [0,16384): z ; [16384,147456): w
  float* ws  = (float*)d_ws;

  float* G  = ws;                                   // 32*8*512   = 131072 f32
  float* SC = G  + (size_t)BX * HEADSX * DIMX;      // 32*4096*8  = 1048576 f32
  float* AT = SC + (size_t)BX * SX * HEADSX;        // 32*8*4096  = 1048576 f32
  float* U  = AT + (size_t)BX * HEADSX * SX;        // 32*8*512   = 131072 f32

  ca_qg_kernel     <<<BX,              256, 0, stream>>>(q, w_in, b_in, G);
  ca_scores_kernel <<<BX * 16,         256, 0, stream>>>(k, G, SC);
  ca_softmax_kernel<<<BX * HEADSX,     256, 0, stream>>>(SC, AT);
  ca_wmean_kernel  <<<BX * 16,         256, 0, stream>>>(AT, out + BX * DIMX);
  ca_ctx_kernel    <<<BX * 8,          128, 0, stream>>>(v, AT, U);
  ca_outproj_kernel<<<BX,              256, 0, stream>>>(U, w_in, b_in, ow, ob, out);
}
